// SSIMLoss_90134183673886
// MI455X (gfx1250) — matrix operand order repaired and checked by reading, not compiled, not run on hardware
//
#include <hip/hip_runtime.h>

typedef _Float16 v16h __attribute__((ext_vector_type(16)));
typedef _Float16 v8h  __attribute__((ext_vector_type(8)));
typedef float    v8f  __attribute__((ext_vector_type(8)));

#define C1F 0.0001f   /* 0.01^2 */
#define C2F 0.0009f   /* 0.03^2 */

// 11-tap Gaussian, sigma=1.5, normalized (matches reference _gaussian_window)
__constant__ float c_w[11] = {
    0.00102839f, 0.00759876f, 0.03600077f, 0.10936071f, 0.21300553f,
    0.26601172f, 0.21300553f, 0.10936071f, 0.03600077f, 0.00759876f,
    0.00102839f
};

__device__ __forceinline__ float wsel(int d) {
    bool in = (unsigned)d <= 10u;
    int dc = in ? d : 0;          // avoid OOB constant read
    return in ? c_w[dc] : 0.0f;
}

// K index held by element e of the 16-bit A-matrix (16x32) for lane-half h.
// ISA layout: half 0 -> K 0..7 (v0..3), 16..23 (v4..7); half 1 -> +8.
__device__ __forceinline__ int aK(int e, int h) {
    return e + h * 8 + (e >= 8 ? 8 : 0);
}

__device__ __forceinline__ v8f wmma16(v16h a, v16h b, v8f c) {
    return __builtin_amdgcn_wmma_f32_16x16x32_f16(
        /*neg_a=*/false, a, /*neg_b=*/false, b,
        /*c_mod=*/(short)0, c, /*reuse_a=*/false, /*reuse_b=*/false);
}

__global__ void ssim_init_kernel(float* acc) { acc[0] = 0.0f; }

__global__ void ssim_final_kernel(float* out, const float* acc) {
    // mean over 16*3*512*512 = 12582912 elements
    out[0] = 1.0f - acc[0] * (1.0f / 12582912.0f);
}

__device__ __forceinline__ void unpack16(float* d, float4 a, float4 b,
                                         float4 c, float4 e) {
    d[0] = a.x;  d[1] = a.y;  d[2] = a.z;  d[3] = a.w;
    d[4] = b.x;  d[5] = b.y;  d[6] = b.z;  d[7] = b.w;
    d[8] = c.x;  d[9] = c.y;  d[10] = c.z; d[11] = c.w;
    d[12] = e.x; d[13] = e.y; d[14] = e.z; d[15] = e.w;
}

__global__ __launch_bounds__(256)
void ssim_main_kernel(const float* __restrict__ img1,
                      const float* __restrict__ img2,
                      float* __restrict__ acc) {
    // One wave = one 16x16 output tile; 8 waves per block.
    // Per-wave LDS panel, column-major [N][K]: 16 columns x 5 v8h slots
    // (slots 0..3 hold K=0..31 as f16, slot 4 = pad -> 80B column stride,
    // conflict-free b128 banking, 1280 B per wave).
    __shared__ v8h ldsB[8][16 * 5];

    const int lane = threadIdx.x & 31;
    const int wid  = threadIdx.x >> 5;
    const int half = lane >> 4;
    const int mn   = lane & 15;   // M for A/C/D layouts, N for B layout

    // Banded weight matrices.
    // Horizontal (B operand, 32x16): window starts at x0-8, so
    //   out[N] = sum_K in[x0-8+K] * w[K-N-3]  (valid K in [3,28])
    // Vertical (A operand, 16x32): A[M][K] = w[K-M]  (H rows y0-5+K)
    v16h whB, wvA;
#pragma unroll
    for (int e = 0; e < 16; ++e) {
        int Kb = aK(e, half);                 // B layout K index
        whB[e] = (_Float16)wsel(Kb - mn - 3);
        int Ka = e + half * 16;               // A layout K index
        wvA[e] = (_Float16)wsel(Ka - mn);
    }

    // Tile decode: 32x32 tiles per plane, 48 planes, 49152 tiles total.
    const int t  = blockIdx.x * 8 + wid;
    const int tx = t & 31;
    const int ty = (t >> 5) & 31;
    const int pl = t >> 10;
    const float* __restrict__ p1 = img1 + (size_t)pl * (512 * 512);
    const float* __restrict__ p2 = img2 + (size_t)pl * (512 * 512);
    const int x0 = tx << 4;
    const int y0 = ty << 4;

    // Gather input halo into A-layout registers.
    // Row block a: rows y0-5..y0+10 ; row block b: rows y0+11..y0+26.
    // Columns x0-8+K, K=0..31 (zero weight outside [3,28]).
    float r1a[16], r2a[16], r1b[16], r2b[16];
    const bool interior = (tx >= 1) & (tx <= 30) & (ty >= 1) & (ty <= 30);
    if (interior) {
        // All 16B-aligned: segment starts x0-8+8h, x0-4+8h, x0+8+8h, x0+12+8h.
        const size_t rowA = (size_t)(y0 - 5 + mn) * 512 + (x0 - 8 + 8 * half);
        const size_t rowB = (size_t)(y0 + 11 + mn) * 512 + (x0 - 8 + 8 * half);
        const float4* q1a = (const float4*)(p1 + rowA);
        const float4* q2a = (const float4*)(p2 + rowA);
        const float4* q1b = (const float4*)(p1 + rowB);
        const float4* q2b = (const float4*)(p2 + rowB);
        unpack16(r1a, q1a[0], q1a[1], q1a[4], q1a[5]);
        unpack16(r2a, q2a[0], q2a[1], q2a[4], q2a[5]);
        unpack16(r1b, q1b[0], q1b[1], q1b[4], q1b[5]);
        unpack16(r2b, q2b[0], q2b[1], q2b[4], q2b[5]);
    } else {
#pragma unroll
        for (int e = 0; e < 16; ++e) {
            const int K  = aK(e, half);
            const int gx = x0 - 8 + K;
            const bool okx = (gx >= 0) && (gx < 512);

            const int gy0 = y0 - 5 + mn;
            const bool ok0 = okx && (gy0 >= 0) && (gy0 < 512);
            const size_t i0 = (size_t)gy0 * 512 + gx;
            r1a[e] = ok0 ? p1[i0] : 0.0f;
            r2a[e] = ok0 ? p2[i0] : 0.0f;

            const int gy1 = y0 + 11 + mn;
            const bool ok1 = okx && (gy1 < 512);
            const size_t i1 = (size_t)gy1 * 512 + gx;
            r1b[e] = ok1 ? p1[i1] : 0.0f;
            r2b[e] = ok1 ? p2[i1] : 0.0f;
        }
    }

    // Five fields: x1, x2, x1^2, x2^2, x1*x2. Each: 2 horizontal WMMAs,
    // LDS transpose (C/D layout -> B layout, b128 packed), 1 vertical WMMA.
    v8f res[5];
#pragma unroll
    for (int f = 0; f < 5; ++f) {
        v16h a1, a2;
#pragma unroll
        for (int e = 0; e < 16; ++e) {
            const float x1 = r1a[e], x2 = r2a[e];
            const float va = (f == 0) ? x1 : (f == 1) ? x2 :
                             (f == 2) ? x1 * x1 : (f == 3) ? x2 * x2 : x1 * x2;
            a1[e] = (_Float16)va;
            const float y1 = r1b[e], y2 = r2b[e];
            const float vb = (f == 0) ? y1 : (f == 1) ? y2 :
                             (f == 2) ? y1 * y1 : (f == 3) ? y2 * y2 : y1 * y2;
            a2[e] = (_Float16)vb;
        }

        const v8f z = {};
        const v8f h1 = wmma16(a1, whB, z);   // H rows y0-5 .. y0+10 (K 0..15)
        const v8f h2 = wmma16(a2, whB, z);   // H rows y0+11.. y0+26 (K 16..31;
                                             //  K>=26 has zero Wv weight)
        // C/D vgpr r is (M = r + 8*half, N = mn): contiguous K per lane ->
        // one b128 store per H matrix.
        v8h h1h, h2h;
#pragma unroll
        for (int r = 0; r < 8; ++r) {
            h1h[r] = (_Float16)h1[r];
            h2h[r] = (_Float16)h2[r];
        }
        ldsB[wid][mn * 5 + half]     = h1h;   // halves K = 8*half + r
        ldsB[wid][mn * 5 + 2 + half] = h2h;   // halves K = 16 + 8*half + r
        asm volatile("" ::: "memory");        // same-wave DS is in-order

        // B layout: element e holds K = e + 16*half at column N = mn.
        const v8h lo = ldsB[wid][mn * 5 + 2 * half];
        const v8h hi = ldsB[wid][mn * 5 + 2 * half + 1];
        asm volatile("" ::: "memory");
        const v16h bv = __builtin_shufflevector(
            lo, hi, 0, 1, 2, 3, 4, 5, 6, 7, 8, 9, 10, 11, 12, 13, 14, 15);

        res[f] = wmma16(wvA, bv, z);          // vertical pass: Wv x H
    }

    // SSIM map on the 8 per-lane f32 results, then wave reduce + one atomic.
    float sum = 0.0f;
#pragma unroll
    for (int r = 0; r < 8; ++r) {
        const float mu1 = res[0][r], mu2 = res[1][r];
        const float e11 = res[2][r], e22 = res[3][r], e12 = res[4][r];
        const float m11 = mu1 * mu1, m22 = mu2 * mu2, m12 = mu1 * mu2;
        const float s1 = e11 - m11, s2 = e22 - m22, s12 = e12 - m12;
        const float num = (2.0f * m12 + C1F) * (2.0f * s12 + C2F);
        const float den = (m11 + m22 + C1F) * (s1 + s2 + C2F);
        sum += num / den;
    }
#pragma unroll
    for (int off = 16; off > 0; off >>= 1)
        sum += __shfl_xor(sum, off, 32);
    if (lane == 0)
        atomicAdd(acc, sum);
}

extern "C" void kernel_launch(void* const* d_in, const int* in_sizes, int n_in,
                              void* d_out, int out_size, void* d_ws, size_t ws_size,
                              hipStream_t stream) {
    (void)in_sizes; (void)n_in; (void)out_size; (void)ws_size;
    const float* img1 = (const float*)d_in[0];
    const float* img2 = (const float*)d_in[1];
    float* out = (float*)d_out;
    float* acc = (float*)d_ws;

    ssim_init_kernel<<<1, 1, 0, stream>>>(acc);
    // 49152 tiles (16*3*32*32), 8 waves/block -> 6144 blocks
    ssim_main_kernel<<<dim3(6144), dim3(256), 0, stream>>>(img1, img2, acc);
    ssim_final_kernel<<<1, 1, 0, stream>>>(out, acc);
}
